// RRN_23888608101388
// MI455X (gfx1250) — matrix operand order repaired and checked by reading, not compiled, not run on hardware
//
#include <hip/hip_runtime.h>

// ---------------------------------------------------------------------------
// RRN on MI455X (gfx1250, wave32).
// BF16 WMMA (v_wmma_f32_16x16x32_bf16) for all GEMMs, f32 accumulation,
// hardware f32 atomics for the scatter-add, wave32 shuffle for l2norm.
// M=32 per wave in the K-accumulation phases to double wmma density per
// B-fragment load (weights stream through WGP$/L2; activations live in L2).
// ---------------------------------------------------------------------------

#define RRN_N 100000
#define RRN_D 128
#define RRN_K 64
#define RRN_L 16
#define RRN_E 25000
#define RRN_H 128
#define RRN_ITERS 2

typedef __bf16 v16bf __attribute__((ext_vector_type(16)));
typedef float  v8f   __attribute__((ext_vector_type(8)));

static __device__ __forceinline__ v8f wmma_bf16(v16bf a, v16bf b, v8f c) {
    // 8 args: (neg_a, A, neg_b, B, c_mod, C, reuse_a, reuse_b)
    return __builtin_amdgcn_wmma_f32_16x16x32_bf16(
        false, a, false, b, (short)0, c, false, false);
}

static __device__ __forceinline__ v8f v8f_zero() {
    v8f z = {0.f, 0.f, 0.f, 0.f, 0.f, 0.f, 0.f, 0.f};
    return z;
}

// A-fragment (16x32 bf16, MxK) from a bf16 row pointer with unit-stride K.
// ISA layout: lanes 0-15 hold M=lane, K={0..7,16..23}; lanes 16-31 hold
// M=lane-16, K={8..15,24..31}.  khalf = (lane>>4)*8.
static __device__ __forceinline__ v16bf load_afrag_bf16(const __bf16* row, int khalf) {
    v16bf a;
#pragma unroll
    for (int i = 0; i < 8; ++i) {
        a[i]     = row[khalf + i];
        a[8 + i] = row[16 + khalf + i];
    }
    return a;
}

// Same fragment but converting from f32 source (emits v_cvt_pk_bf16_f32).
static __device__ __forceinline__ v16bf load_afrag_f32(const float* row, int khalf) {
    v16bf a;
#pragma unroll
    for (int i = 0; i < 8; ++i) {
        a[i]     = (__bf16)row[khalf + i];
        a[8 + i] = (__bf16)row[16 + khalf + i];
    }
    return a;
}

// B fragments are pre-packed so each lane reads 16 contiguous bf16 (32B).
static __device__ __forceinline__ v16bf load_bfrag(const __bf16* packed,
                                                   int frag, int lane) {
    return *(const v16bf*)(packed + ((size_t)frag * 32 + lane) * 16);
}

// ---------------------------------------------------------------------------
// Weight repack: W[K][Nn] row-major f32 -> packed bf16 fragments.
// Fragment (kc,nt): 32 lanes x 16 bf16; lane n<16 -> K={0..7,16..23},
// lane n>=16 -> K={8..15,24..31} (within chunk kc), column nt*16+(n&15).
// ---------------------------------------------------------------------------
__global__ void pack_w(const float* __restrict__ W, __bf16* __restrict__ out,
                       int Nn, int nfrag, long inStride, long outStride) {
    int t   = blockIdx.x * blockDim.x + threadIdx.x;
    int mat = blockIdx.y;
    if (t >= nfrag * 32) return;
    int frag = t >> 5, lane = t & 31;
    int Nt = Nn >> 4;
    int kc = frag / Nt, nt = frag % Nt;
    int n     = nt * 16 + (lane & 15);
    int kbase = kc * 32 + (lane >> 4) * 8;
    const float* Wm = W + (size_t)mat * inStride;
    __bf16* Om = out + (size_t)mat * outStride + ((size_t)frag * 32 + lane) * 16;
#pragma unroll
    for (int e = 0; e < 16; ++e)
        Om[e] = (__bf16)Wm[(size_t)(kbase + (e < 8 ? e : e + 8)) * Nn + n];
}

__global__ void zero_k(float4* __restrict__ p, int n4) {
    int i = blockIdx.x * blockDim.x + threadIdx.x;
    if (i < n4) p[i] = make_float4(0.f, 0.f, 0.f, 0.f);
}

// ---------------------------------------------------------------------------
// ClassUpdate: upd = relu(concat(e, memb) @ W1 + b1) @ W2 + b2.
// One wave per 32-row tile (N = 3125 * 32 exactly); M=32 in phase 1,
// two 16-row halves in phase 2.
// ---------------------------------------------------------------------------
__global__ __launch_bounds__(32) void class_update_k(
    const float* __restrict__ e, const float* __restrict__ memb,
    const __bf16* __restrict__ W1p, const float* __restrict__ b1,
    const __bf16* __restrict__ W2p, const float* __restrict__ b2,
    float* __restrict__ upd, __bf16* __restrict__ updbf) {
    __shared__ __bf16 hT[32 * 128];
    int lane  = threadIdx.x;
    int tile  = blockIdx.x;
    int m     = lane & 15;
    int half  = lane >> 4;
    int khalf = half * 8;
    size_t row0 = (size_t)tile * 32 + m;       // A-frag 0 rows
    size_t row1 = row0 + 16;                   // A-frag 1 rows

    v8f accH[16];
#pragma unroll
    for (int i = 0; i < 16; ++i) accH[i] = v8f_zero();

    // GEMM1: [32 x 192] x [192 x 128], K = 6 chunks of 32.
#pragma unroll
    for (int kc = 0; kc < 6; ++kc) {
        int col0 = kc * 32;
        const float *ar0, *ar1;
        int co;
        if (col0 < RRN_D) {
            ar0 = e + row0 * RRN_D; ar1 = e + row1 * RRN_D; co = col0;
        } else {
            ar0 = memb + row0 * RRN_K; ar1 = memb + row1 * RRN_K; co = col0 - RRN_D;
        }
        v16bf a0 = load_afrag_f32(ar0 + co, khalf);
        v16bf a1 = load_afrag_f32(ar1 + co, khalf);
#pragma unroll
        for (int nt = 0; nt < 8; ++nt) {
            v16bf b = load_bfrag(W1p, kc * 8 + nt, lane);
            accH[nt]     = wmma_bf16(a0, b, accH[nt]);
            accH[8 + nt] = wmma_bf16(a1, b, accH[8 + nt]);
        }
    }
    // bias + relu -> LDS (C layout: VGPR r holds M=r+8*half, N=nt*16+(lane&15))
#pragma unroll
    for (int g = 0; g < 2; ++g) {
#pragma unroll
        for (int nt = 0; nt < 8; ++nt) {
            int n = nt * 16 + m;
            float bn = b1[n];
#pragma unroll
            for (int r = 0; r < 8; ++r) {
                int mm = g * 16 + r + half * 8;
                hT[mm * 128 + n] = (__bf16)fmaxf(accH[g * 8 + nt][r] + bn, 0.f);
            }
        }
    }
    __syncthreads();

    // GEMM2: [32 x 128] x [128 x 128]; loop the two 16-row halves.
#pragma unroll
    for (int g = 0; g < 2; ++g) {
        v8f accU[8];
#pragma unroll
        for (int nt = 0; nt < 8; ++nt) accU[nt] = v8f_zero();
#pragma unroll
        for (int kc = 0; kc < 4; ++kc) {
            v16bf a = load_afrag_bf16(hT + (g * 16 + m) * 128 + kc * 32, khalf);
#pragma unroll
            for (int nt = 0; nt < 8; ++nt) {
                v16bf b = load_bfrag(W2p, kc * 8 + nt, lane);
                accU[nt] = wmma_bf16(a, b, accU[nt]);
            }
        }
#pragma unroll
        for (int nt = 0; nt < 8; ++nt) {
            int n = nt * 16 + m;
            float bn = b2[n];
#pragma unroll
            for (int r = 0; r < 8; ++r) {
                int mm = g * 16 + r + half * 8;
                size_t orow = (size_t)tile * 32 + mm;
                float v = accU[nt][r] + bn;
                upd[orow * RRN_D + n]   = v;
                updbf[orow * RRN_D + n] = (__bf16)v;
            }
        }
    }
}

// ---------------------------------------------------------------------------
// RelationUpdate layer: gather -> MLP (3 GEMMs) -> atomic scatter-add.
// grid = (ceil(E/32), L), one wave per (layer, 32-edge tile).
// ---------------------------------------------------------------------------
__global__ __launch_bounds__(32) void relation_k(
    const __bf16* __restrict__ updbf,
    const __bf16* __restrict__ W1p, const float* __restrict__ b1,
    const __bf16* __restrict__ Wsp, const float* __restrict__ bsv,
    const __bf16* __restrict__ Wop, const float* __restrict__ bov,
    const int* __restrict__ sIdx, const int* __restrict__ oIdx,
    float* __restrict__ accv) {
    __shared__ __bf16 hT[32 * 128];
    __shared__ int sI[32];
    __shared__ int oI[32];
    int lane  = threadIdx.x;
    int l     = blockIdx.y;
    int e0    = blockIdx.x * 32;
    int m     = lane & 15;
    int half  = lane >> 4;
    int khalf = half * 8;

    {
        int ee = e0 + lane;
        if (ee >= RRN_E) ee = RRN_E - 1;
        sI[lane] = sIdx[(size_t)l * RRN_E + ee];
        oI[lane] = oIdx[(size_t)l * RRN_E + ee];
    }
    __syncthreads();
    const __bf16* srow0 = updbf + (size_t)sI[m] * RRN_D;
    const __bf16* orow0 = updbf + (size_t)oI[m] * RRN_D;
    const __bf16* srow1 = updbf + (size_t)sI[16 + m] * RRN_D;
    const __bf16* orow1 = updbf + (size_t)oI[16 + m] * RRN_D;
    const __bf16* W1l = W1p + (size_t)l * (64 * 512);
    const __bf16* Wsl = Wsp + (size_t)l * (32 * 512);
    const __bf16* Wol = Wop + (size_t)l * (32 * 512);

    // h = relu(pair @ W1_l + b1_l); pair = concat(e_s, e_o): K = 8 chunks,
    // M = 32 (two A fragments sharing each B fragment).
    v8f accH[16];
#pragma unroll
    for (int i = 0; i < 16; ++i) accH[i] = v8f_zero();
#pragma unroll
    for (int kc = 0; kc < 8; ++kc) {
        const __bf16* base0 = (kc < 4) ? (srow0 + kc * 32) : (orow0 + (kc - 4) * 32);
        const __bf16* base1 = (kc < 4) ? (srow1 + kc * 32) : (orow1 + (kc - 4) * 32);
        v16bf a0 = load_afrag_bf16(base0, khalf);
        v16bf a1 = load_afrag_bf16(base1, khalf);
#pragma unroll
        for (int nt = 0; nt < 8; ++nt) {
            v16bf b = load_bfrag(W1l, kc * 8 + nt, lane);
            accH[nt]     = wmma_bf16(a0, b, accH[nt]);
            accH[8 + nt] = wmma_bf16(a1, b, accH[8 + nt]);
        }
    }
#pragma unroll
    for (int g = 0; g < 2; ++g) {
#pragma unroll
        for (int nt = 0; nt < 8; ++nt) {
            int n = nt * 16 + m;
            float bn = b1[(size_t)l * RRN_H + n];
#pragma unroll
            for (int r = 0; r < 8; ++r) {
                int mm = g * 16 + r + half * 8;
                hT[mm * 128 + n] = (__bf16)fmaxf(accH[g * 8 + nt][r] + bn, 0.f);
            }
        }
    }
    __syncthreads();

    // us = h @ Ws_l, uo = h @ Wo_l : K = 4 chunks, looped over 16-row halves.
#pragma unroll
    for (int g = 0; g < 2; ++g) {
        v8f accS[8], accO[8];
#pragma unroll
        for (int nt = 0; nt < 8; ++nt) { accS[nt] = v8f_zero(); accO[nt] = v8f_zero(); }
#pragma unroll
        for (int kc = 0; kc < 4; ++kc) {
            v16bf a = load_afrag_bf16(hT + (g * 16 + m) * 128 + kc * 32, khalf);
#pragma unroll
            for (int nt = 0; nt < 8; ++nt) {
                v16bf bs = load_bfrag(Wsl, kc * 8 + nt, lane);
                accS[nt] = wmma_bf16(a, bs, accS[nt]);
                v16bf bo = load_bfrag(Wol, kc * 8 + nt, lane);
                accO[nt] = wmma_bf16(a, bo, accO[nt]);
            }
        }
        // scatter-add (hardware f32 atomics), masking the partial last tile.
#pragma unroll
        for (int nt = 0; nt < 8; ++nt) {
            int n = nt * 16 + m;
            float bsn = bsv[(size_t)l * RRN_D + n];
            float bon = bov[(size_t)l * RRN_D + n];
#pragma unroll
            for (int r = 0; r < 8; ++r) {
                int mm = g * 16 + r + half * 8;
                if (e0 + mm < RRN_E) {
                    __hip_atomic_fetch_add(&accv[(size_t)sI[mm] * RRN_D + n],
                                           accS[nt][r] + bsn,
                                           __ATOMIC_RELAXED, __HIP_MEMORY_SCOPE_AGENT);
                    __hip_atomic_fetch_add(&accv[(size_t)oI[mm] * RRN_D + n],
                                           accO[nt][r] + bon,
                                           __ATOMIC_RELAXED, __HIP_MEMORY_SCOPE_AGENT);
                }
            }
        }
    }
}

// ---------------------------------------------------------------------------
// e = l2norm(upd + acc). 8 waves per block, one row per wave; wave32
// xor-shuffle reduction.
// ---------------------------------------------------------------------------
__global__ __launch_bounds__(256) void finalize_k(const float* __restrict__ upd,
                                                  const float* __restrict__ accv,
                                                  float* __restrict__ outp) {
    size_t row = (size_t)blockIdx.x * 8 + (threadIdx.x >> 5);
    int lane = threadIdx.x & 31;
    size_t base = row * RRN_D + lane * 4;
    float4 uu = *(const float4*)(upd + base);
    float4 aa = *(const float4*)(accv + base);
    float x0 = uu.x + aa.x, x1 = uu.y + aa.y, x2 = uu.z + aa.z, x3 = uu.w + aa.w;
    float ss = x0 * x0 + x1 * x1 + x2 * x2 + x3 * x3;
#pragma unroll
    for (int off = 16; off >= 1; off >>= 1) ss += __shfl_xor(ss, off, 32);
    float inv = 1.0f / fmaxf(sqrtf(ss), 1e-12f);
    float4 o = make_float4(x0 * inv, x1 * inv, x2 * inv, x3 * inv);
    *(float4*)(outp + base) = o;
}

// ---------------------------------------------------------------------------
extern "C" void kernel_launch(void* const* d_in, const int* in_sizes, int n_in,
                              void* d_out, int out_size, void* d_ws, size_t ws_size,
                              hipStream_t stream) {
    (void)in_sizes; (void)n_in; (void)out_size; (void)ws_size;
    const float* emb  = (const float*)d_in[0];
    const float* memb = (const float*)d_in[1];
    const float* cW1  = (const float*)d_in[2];
    const float* cb1  = (const float*)d_in[3];
    const float* cW2  = (const float*)d_in[4];
    const float* cb2  = (const float*)d_in[5];
    const float* rW1  = (const float*)d_in[6];
    const float* rb1  = (const float*)d_in[7];
    const float* rWs  = (const float*)d_in[8];
    const float* rbs  = (const float*)d_in[9];
    const float* rWo  = (const float*)d_in[10];
    const float* rbo  = (const float*)d_in[11];
    const int*   sIdx = (const int*)d_in[12];
    const int*   oIdx = (const int*)d_in[13];

    char* ws = (char*)d_ws;
    size_t off = 0;
    auto take = [&](size_t bytes) -> void* {
        void* p = ws + off;
        off += (bytes + 255) & ~(size_t)255;
        return p;
    };
    const size_t ND = (size_t)RRN_N * RRN_D;
    float*  upd   = (float*)take(ND * 4);                      // 51.2 MB
    float*  accb  = (float*)take(ND * 4);                      // 51.2 MB (acc / e ping-pong)
    __bf16* updbf = (__bf16*)take(ND * 2);                     // 25.6 MB (L2-resident gather table)
    __bf16* W1p   = (__bf16*)take((size_t)192 * 128 * 2);
    __bf16* W2p   = (__bf16*)take((size_t)128 * 128 * 2);
    __bf16* rW1p  = (__bf16*)take((size_t)RRN_L * 256 * 128 * 2);
    __bf16* rWsp  = (__bf16*)take((size_t)RRN_L * 128 * 128 * 2);
    __bf16* rWop  = (__bf16*)take((size_t)RRN_L * 128 * 128 * 2);

    // Repack weights to fragment-native bf16 (tiny, once per launch).
    pack_w<<<dim3(6, 1), 256, 0, stream>>>(cW1, W1p, 128, 48, 0, 0);
    pack_w<<<dim3(4, 1), 256, 0, stream>>>(cW2, W2p, 128, 32, 0, 0);
    pack_w<<<dim3(8, RRN_L), 256, 0, stream>>>(rW1, rW1p, 128, 64,
                                               256L * 128, 64L * 512);
    pack_w<<<dim3(4, RRN_L), 256, 0, stream>>>(rWs, rWsp, 128, 32,
                                               128L * 128, 32L * 512);
    pack_w<<<dim3(4, RRN_L), 256, 0, stream>>>(rWo, rWop, 128, 32,
                                               128L * 128, 32L * 512);

    const int nTilesN = RRN_N / 32;               // 3125 (exact)
    const int nTilesE = (RRN_E + 31) / 32;        // 782
    for (int t = 0; t < RRN_ITERS; ++t) {
        const float* eIn = (t == 0) ? emb : accb;
        class_update_k<<<nTilesN, 32, 0, stream>>>(eIn, memb, W1p, cb1, W2p, cb2,
                                                   upd, updbf);
        zero_k<<<(int)(ND / 4 / 256), 256, 0, stream>>>((float4*)accb, (int)(ND / 4));
        relation_k<<<dim3(nTilesE, RRN_L), 32, 0, stream>>>(updbf, rW1p, rb1,
                                                            rWsp, rbs, rWop, rbo,
                                                            sIdx, oIdx, accb);
        float* outp = (t == RRN_ITERS - 1) ? (float*)d_out : accb;
        finalize_k<<<12500, 256, 0, stream>>>(upd, accb, outp);
    }
}
